// Encoder_58772332478696
// MI455X (gfx1250) — matrix operand order repaired and checked by reading, not compiled, not run on hardware
//
#include <hip/hip_runtime.h>
#include <hip/hip_bf16.h>

typedef __attribute__((ext_vector_type(16))) __bf16 v16bf;
typedef __attribute__((ext_vector_type(8)))  __bf16 v8bf;
typedef __attribute__((ext_vector_type(8)))  float  v8f;

#define L_SEQ 256
#define B_SZ  80
#define DIN   512
#define HDIM  1024
#define N3    3072
#define NT3   192           // N3/16 column tiles
#define MROWS (L_SEQ*B_SZ)  // 20480
#define NWG_SCAN 40         // 40 WGs * 8 waves = 320 waves = 5 mtiles * 64 ntiles

union ABu { v16bf v; v8bf h[2]; };

// A fragment (16x32 bf16, MxK), contiguous k order matching k_pack: lanes0-15 M=0..15, half selects K-halves.
// element j: K = kk*32 + half*16 + j  -> two contiguous 16B loads.
__device__ __forceinline__ v16bf load_a_frag(const __bf16* A, int lda, int row_base,
                                             int kk, int lane, int rev) {
  int r = row_base + (lane & 15);
  if (rev) { int t = r / B_SZ, b = r % B_SZ; r = (L_SEQ - 1 - t) * B_SZ + b; }
  int half = lane >> 4;
  const __bf16* p = A + (long)r * lda + kk * 32 + half * 16;
  ABu u;
  u.h[0] = *(const v8bf*)(p);
  u.h[1] = *(const v8bf*)(p + 8);
  return u.v;
}

__device__ __forceinline__ v8f wmma_bf16(v16bf a, v16bf b, v8f c) {
  return __builtin_amdgcn_wmma_f32_16x16x32_bf16(false, a, false, b, (short)0, c,
                                                 false, false);
}

// ---------------- init: zero grid-barrier counters ----------------
__global__ void k_init(int* ctr, int n) {
  int i = blockIdx.x * blockDim.x + threadIdx.x;
  if (i < n) ctr[i] = 0;
}

// ---------------- pack W[K,3072] fp32 -> bf16 B-fragment layout ----------------
// B frag (32x16 KxN): lane holds column n = ntile*16 + (lane&15),
// element j holds K = kk*32 + (lane>>4)*16 + j. Stored so a wave reads
// 32 lanes * 16 bf16 contiguous (fully coalesced b128 pairs).
__global__ void k_pack(const float* W, __bf16* dst, int Kdim) {
  int idx = blockIdx.x * blockDim.x + threadIdx.x;
  int KK = Kdim >> 5;
  int total = NT3 * KK * 32;
  if (idx >= total) return;
  int lane = idx & 31;
  int blk = idx >> 5;
  int kk = blk % KK;
  int ntile = blk / KK;
  int n = ntile * 16 + (lane & 15);
  int kbase = kk * 32 + (lane >> 4) * 16;
  v16bf tmp;
#pragma unroll
  for (int j = 0; j < 16; j++) tmp[j] = (__bf16)W[(long)(kbase + j) * N3 + n];
  ((v16bf*)dst)[idx] = tmp;
}

// ---------------- fp32 -> bf16 elementwise ----------------
__global__ void k_cvt(const float* src, __bf16* dst, long n) {
  long i = (long)blockIdx.x * blockDim.x + threadIdx.x;
  if (i < n) dst[i] = (__bf16)src[i];
}

// ---------------- big input-projection GEMM: out[M,3072] = A[M,K]@Wp + bias --------
// Per wave: 2 M-tiles x 4 N-tiles (A-frag reuse x4, B-frag reuse x2).
__global__ __launch_bounds__(256) void k_gemm_x(const __bf16* A, int Kdim, int rev,
                                                const __bf16* Bp, const float* bias,
                                                float* out) {
  int w = blockIdx.x * 8 + (threadIdx.x >> 5);
  int lane = threadIdx.x & 31;
  int tmi = w % (MROWS / 32);   // 640 row-pair tiles
  int tni = w / (MROWS / 32);   // 48 groups of 4 column tiles
  int KK = Kdim >> 5;
  const v16bf* Bv = (const v16bf*)Bp;
  v8f acc[2][4] = {};
  for (int kk = 0; kk < KK; kk++) {
    v16bf a0 = load_a_frag(A, Kdim, tmi * 32, kk, lane, rev);
    v16bf a1 = load_a_frag(A, Kdim, tmi * 32 + 16, kk, lane, rev);
#pragma unroll
    for (int ni = 0; ni < 4; ni++) {
      v16bf b = Bv[((long)((tni * 4 + ni) * KK + kk)) * 32 + lane];
      acc[0][ni] = wmma_bf16(a0, b, acc[0][ni]);
      acc[1][ni] = wmma_bf16(a1, b, acc[1][ni]);
    }
  }
  int half = lane >> 4;
#pragma unroll
  for (int mi = 0; mi < 2; mi++) {
#pragma unroll
    for (int ni = 0; ni < 4; ni++) {
      int c = (tni * 4 + ni) * 16 + (lane & 15);
      float bb = bias[c];
#pragma unroll
      for (int i = 0; i < 8; i++) {
        int r = tmi * 32 + mi * 16 + half * 8 + i;
        out[(long)r * N3 + c] = acc[mi][ni][i] + bb;
      }
    }
  }
}

// ---------------- device-scope grid barrier (one per timestep) ----------------
__device__ __forceinline__ void grid_barrier(int* c) {
  __builtin_amdgcn_fence(__ATOMIC_RELEASE, "agent");
  __syncthreads();
  if (threadIdx.x == 0) {
    __hip_atomic_fetch_add(c, 1, __ATOMIC_RELEASE, __HIP_MEMORY_SCOPE_AGENT);
    while (__hip_atomic_load(c, __ATOMIC_ACQUIRE, __HIP_MEMORY_SCOPE_AGENT) < NWG_SCAN) {
      __builtin_amdgcn_s_sleep(1);
    }
  }
  __syncthreads();
  __builtin_amdgcn_fence(__ATOMIC_ACQUIRE, "agent");
}

// ---------------- persistent GRU scan (one direction per launch) ----------------
// Each wave owns one (mtile, ntile) of the [80,1024] state; it computes the
// z/r/h gate tiles (columns n, n+H, n+2H) of h@Wh with a SHARED A fragment,
// then the full elementwise update locally -> only one barrier per step.
__global__ __launch_bounds__(256) void k_gru_scan(const float* xproj, const float* mask,
                                                  const __bf16* Whp_, const float* bh,
                                                  float* hbuf, __bf16* hbufb,
                                                  __bf16* right_b, float* outbuf,
                                                  int* ctr, int backward) {
  int w = blockIdx.x * 8 + (threadIdx.x >> 5);  // 0..319
  int lane = threadIdx.x & 31;
  int mtile = w % 5;
  int ntile = w / 5;  // 0..63
  int half = lane >> 4;
  int n = ntile * 16 + (lane & 15);
  const v16bf* Wv = (const v16bf*)Whp_;
  float bz = bh[n], br = bh[n + HDIM], bhh = bh[n + 2 * HDIM];

  for (int t = 0; t < L_SEQ; t++) {
    if (t > 0) grid_barrier(ctr + (t - 1));
    int cur = t & 1, prv = cur ^ 1;
    v8f accZ = {}, accR = {}, accH = {};
    if (t > 0) {  // uniform branch: EXEC stays all-ones for WMMA
      const __bf16* hA = hbufb + prv * B_SZ * HDIM;
      for (int kk = 0; kk < HDIM / 32; kk++) {
        v16bf a  = load_a_frag(hA, HDIM, mtile * 16, kk, lane, 0);
        v16bf b0 = Wv[((long)((ntile        ) * 32 + kk)) * 32 + lane];
        v16bf b1 = Wv[((long)((ntile +  64) * 32 + kk)) * 32 + lane];
        v16bf b2 = Wv[((long)((ntile + 128) * 32 + kk)) * 32 + lane];
        accZ = wmma_bf16(a, b0, accZ);
        accR = wmma_bf16(a, b1, accR);
        accH = wmma_bf16(a, b2, accH);
      }
    }
    const float* hprev = hbuf + prv * B_SZ * HDIM;
    float* hcur = hbuf + cur * B_SZ * HDIM;
    __bf16* hcurb = hbufb + cur * B_SZ * HDIM;
    int tout = backward ? (L_SEQ - 1 - t) : t;  // time index for mask/output
#pragma unroll
    for (int i = 0; i < 8; i++) {
      int m = mtile * 16 + half * 8 + i;  // batch row
      long xrow = (long)(t * B_SZ + m) * N3;
      float xz = xproj[xrow + n];
      float xr = xproj[xrow + HDIM + n];
      float xh = xproj[xrow + 2 * HDIM + n];
      float hp = (t > 0) ? hprev[m * HDIM + n] : 0.0f;
      float z = 1.0f / (1.0f + __expf(-(xz + accZ[i] + bz)));
      float r = 1.0f / (1.0f + __expf(-(xr + accR[i] + br)));
      float ht = tanhf(xh + r * (accH[i] + bhh));
      float mm = mask[tout * B_SZ + m];
      float hn = (1.0f - z) * hp + z * ht;
      hn = mm * hn + (1.0f - mm) * hp;
      hcur[m * HDIM + n] = hn;
      hcurb[m * HDIM + n] = (__bf16)hn;
      if (!backward) right_b[(long)(t * B_SZ + m) * HDIM + n] = (__bf16)hn;
      else           outbuf[(long)(tout * B_SZ + m) * HDIM + n] = hn;
    }
  }
}

extern "C" void kernel_launch(void* const* d_in, const int* in_sizes, int n_in,
                              void* d_out, int out_size, void* d_ws, size_t ws_size,
                              hipStream_t stream) {
  const float* xs   = (const float*)d_in[0];
  const float* mask = (const float*)d_in[1];
  const float* fWx  = (const float*)d_in[2];
  const float* fWh  = (const float*)d_in[3];
  const float* fbx  = (const float*)d_in[4];
  const float* fbh  = (const float*)d_in[5];
  const float* bWx  = (const float*)d_in[6];
  const float* bWh  = (const float*)d_in[7];
  const float* bbx  = (const float*)d_in[8];
  const float* bbh  = (const float*)d_in[9];
  float* out = (float*)d_out;
  (void)in_sizes; (void)n_in; (void)out_size; (void)ws_size;

  char* ws = (char*)d_ws;
  size_t off = 0;
  auto alloc = [&](size_t bytes) {
    void* p = ws + off;
    off = (off + bytes + 255) & ~(size_t)255;
    return p;
  };
  int*    ctr     = (int*)   alloc(512 * sizeof(int));
  __bf16* fWxp    = (__bf16*)alloc((size_t)DIN  * N3 * 2);
  __bf16* fWhp    = (__bf16*)alloc((size_t)HDIM * N3 * 2);
  __bf16* bWxp    = (__bf16*)alloc((size_t)HDIM * N3 * 2);
  __bf16* bWhp    = (__bf16*)alloc((size_t)HDIM * N3 * 2);
  __bf16* xs_b    = (__bf16*)alloc((size_t)MROWS * DIN * 2);
  __bf16* right_b = (__bf16*)alloc((size_t)MROWS * HDIM * 2);
  float*  hbuf    = (float*) alloc((size_t)2 * B_SZ * HDIM * 4);
  __bf16* hbufb   = (__bf16*)alloc((size_t)2 * B_SZ * HDIM * 2);
  float*  xproj   = (float*) alloc((size_t)MROWS * N3 * 4);

  k_init<<<2, 256, 0, stream>>>(ctr, 512);

  int tot512  = NT3 * (DIN  / 32) * 32;  // 98304
  int tot1024 = NT3 * (HDIM / 32) * 32;  // 196608
  k_pack<<<(tot512  + 255) / 256, 256, 0, stream>>>(fWx, fWxp, DIN);
  k_pack<<<(tot1024 + 255) / 256, 256, 0, stream>>>(fWh, fWhp, HDIM);
  k_pack<<<(tot1024 + 255) / 256, 256, 0, stream>>>(bWx, bWxp, HDIM);
  k_pack<<<(tot1024 + 255) / 256, 256, 0, stream>>>(bWh, bWhp, HDIM);

  long nxs = (long)MROWS * DIN;
  k_cvt<<<(int)((nxs + 255) / 256), 256, 0, stream>>>(xs, xs_b, nxs);

  // forward input projection + forward scan
  k_gemm_x<<<3840, 256, 0, stream>>>(xs_b, DIN, 0, fWxp, fbx, xproj);
  k_gru_scan<<<NWG_SCAN, 256, 0, stream>>>(xproj, mask, fWhp, fbh, hbuf, hbufb,
                                           right_b, nullptr, ctr, 0);
  // backward input projection (time-reversed A rows) + backward scan
  k_gemm_x<<<3840, 256, 0, stream>>>(right_b, HDIM, 1, bWxp, bbx, xproj);
  k_gru_scan<<<NWG_SCAN, 256, 0, stream>>>(xproj, mask, bWhp, bbh, hbuf, hbufb,
                                           nullptr, out, ctr + 256, 1);
}
